// AudioToMelSpectrogramPreprocessor_56727928046315
// MI455X (gfx1250) — matrix operand order repaired and checked by reading, not compiled, not run on hardware
//
#include <hip/hip_runtime.h>
#include <hip/hip_bf16.h>
#include <math.h>

// ---------------------------------------------------------------------------
// AudioToMelSpectrogramPreprocessor for MI455X (gfx1250, wave32, WMMA).
// DFT and mel projection mapped onto v_wmma_f32_16x16x32_f16.
// N-tiles processed in groups of 3 so accumulators stay pinned in their WMMA
// D-ranges (no v_mov/v_nop shuffling), EXEC all-ones by construction.
// ---------------------------------------------------------------------------

typedef __attribute__((ext_vector_type(16))) _Float16 v16h;
typedef __attribute__((ext_vector_type(8)))  _Float16 v8h;
typedef __attribute__((ext_vector_type(8)))  float    v8f;

#define NB        32
#define LSIG      256000
#define HOP       160
#define NFFT      512
#define NMEL      64
#define NFREQ     257          // NFFT/2+1
#define T_FR      1601         // LSIG/HOP + 1
#define T_PAD     1616         // padded to multiple of 16
#define NTILE_T   101          // ceil(1601/16)
#define XP_LEN    (LSIG + NFFT)            // 256512
#define COLS_COS  272          // 257 padded to 16
#define NT_TILES  36           // (272 cos + 272 sin) padded to 576 = 36*16
#define KT_DFT    16           // 512/32
#define KT_MEL    9            // 288/32 (257 padded to 288)
#define LDS_SAMP  2912         // 15*160 + 512
#define WF_STRIDE 520          // 512 + 8 halfs: distinct LDS banks across rows
#define SPEC_STR  577          // 576 + 1 floats: distinct LDS banks across rows
#define LOG_GUARD 5.9604644775390625e-8f   // 2^-24

// Fragment index maps (wave32, 16-bit operands, ISA 7.12.2).
__device__ __forceinline__ int kmapA(int lanegrp, int h) {
  return lanegrp == 0 ? (h < 8 ? h : h + 8) : (h < 8 ? h + 8 : h + 16);
}
__device__ __forceinline__ int kmapB(int lanegrp, int h) {
  return kmapA(lanegrp, h);
}

// ---------------------------------------------------------------------------
// Init: DFT basis as pre-swizzled B fragments.
// dftp[((kt*36 + nt)*32 + lane)*16 + h];  col<272 -> cos, col in [272,544) -> sin.
// ---------------------------------------------------------------------------
__global__ void init_dft(_Float16* __restrict__ dftp) {
  int e = blockIdx.x * 256 + threadIdx.x;
  if (e >= KT_DFT * NT_TILES * 32 * 16) return;
  int h    = e & 15;
  int lane = (e >> 4) & 31;
  int nt   = (e >> 9) % NT_TILES;
  int kt   = e / (512 * NT_TILES);
  int k    = 32 * kt + kmapB(lane >> 4, h);
  int col  = nt * 16 + (lane & 15);
  float v  = 0.0f;
  if (col < 2 * COLS_COS) {
    int f = (col < COLS_COS) ? col : col - COLS_COS;
    if (f < NFREQ) {
      int ph = (f * k) & (NFFT - 1);       // exact phase mod 512
      float ang = 6.283185307179586f * (float)ph / (float)NFFT;
      v = (col < COLS_COS) ? cosf(ang) : sinf(ang);
    }
  }
  dftp[e] = (_Float16)v;
}

// Mel filterbank as pre-swizzled A fragments (f32 -> f16), K padded 257 -> 288.
__global__ void init_fb(const float* __restrict__ fb, _Float16* __restrict__ fbp) {
  int e = blockIdx.x * 256 + threadIdx.x;
  if (e >= 4 * KT_MEL * 32 * 16) return;
  int h    = e & 15;
  int lane = (e >> 4) & 31;
  int kt   = (e >> 9) % KT_MEL;
  int mt   = e / (512 * KT_MEL);
  int mel  = 16 * mt + (lane & 15);
  int k    = 32 * kt + kmapA(lane >> 4, h);
  float v  = (k < NFREQ) ? fb[mel * NFREQ + k] : 0.0f;
  fbp[e] = (_Float16)v;
}

// ---------------------------------------------------------------------------
// Stage A: pre-emphasis + reflect pad into xp[NB][XP_LEN].
// ---------------------------------------------------------------------------
__global__ void stageA(const float* __restrict__ x, float* __restrict__ xp) {
  size_t i = (size_t)blockIdx.x * 256 + threadIdx.x;
  if (i >= (size_t)NB * XP_LEN) return;
  int b  = (int)(i / XP_LEN);
  int ii = (int)(i % XP_LEN);
  int p  = ii - NFFT / 2;
  int s  = (p < 0) ? -p : ((p >= LSIG) ? 2 * LSIG - 2 - p : p);
  const float* xb = x + (size_t)b * LSIG;
  xp[i] = (s == 0) ? xb[0] : xb[s] - 0.97f * xb[s - 1];
}

// ---------------------------------------------------------------------------
// Stage B: window + DFT(WMMA) + power + mel(WMMA) + log.  1 block = 16 frames.
// ---------------------------------------------------------------------------
__global__ void __launch_bounds__(128) stageB(
    const float*    __restrict__ xp,
    const float*    __restrict__ win,      // [512] hann padded (harness input)
    const _Float16* __restrict__ dftp,
    const _Float16* __restrict__ fbp,
    float*          __restrict__ logmel)   // [NB][NMEL][T_PAD]
{
  __shared__ _Float16 wf[16 * WF_STRIDE];       // windowed f16 frames (~16.6 KB)
  __shared__ float    spec[16 * SPEC_STR];      // 16 x 576 spectrum (~36.9 KB)

  int blk  = blockIdx.x;
  int b    = blk / NTILE_T;
  int tile = blk % NTILE_T;
  int t0   = tile * 16;

  const float* xpb = xp + (size_t)b * XP_LEN;
  const int base = t0 * HOP;

  // ---- stage windowed frames to LDS once (shared by all 4 waves)
  if (base + LDS_SAMP <= XP_LEN) {              // block-uniform fast path
    for (int e = threadIdx.x * 4; e < 16 * 512; e += 128 * 4) {
      int m = e >> 9, k = e & 511;
      const float4 xs = *(const float4*)(xpb + base + m * HOP + k);
      const float4 ws = *(const float4*)(win + k);
      wf[m * WF_STRIDE + k + 0] = (_Float16)(xs.x * ws.x);
      wf[m * WF_STRIDE + k + 1] = (_Float16)(xs.y * ws.y);
      wf[m * WF_STRIDE + k + 2] = (_Float16)(xs.z * ws.z);
      wf[m * WF_STRIDE + k + 3] = (_Float16)(xs.w * ws.w);
    }
  } else {                                      // last tile: clamp reads
    for (int e = threadIdx.x; e < 16 * 512; e += 128) {
      int m = e >> 9, k = e & 511;
      int idx = base + m * HOP + k;
      if (idx > XP_LEN - 1) idx = XP_LEN - 1;
      wf[m * WF_STRIDE + k] = (_Float16)(xpb[idx] * win[k]);
    }
  }
  __syncthreads();

  const int lane = threadIdx.x & 31;
  const int w    = threadIdx.x >> 5;            // wave id 0..3
  const int lg   = lane >> 4;
  const int m    = lane & 15;                   // A row / B col within tile

  // ---- DFT: frames[16x512] x basis[512x576]; wave owns 9 N-tiles (3 groups of 3)
  const int aoff = lg ? 8 : 0;                  // kmapA as two contiguous chunks
#pragma unroll
  for (int g = 0; g < 3; ++g) {
    const int nt0 = w + 4 * (3 * g + 0);        // wave-uniform tile ids
    const int nt1 = w + 4 * (3 * g + 1);
    const int nt2 = w + 4 * (3 * g + 2);
    v8f a0 = (v8f){0, 0, 0, 0, 0, 0, 0, 0};
    v8f a1 = (v8f){0, 0, 0, 0, 0, 0, 0, 0};
    v8f a2 = (v8f){0, 0, 0, 0, 0, 0, 0, 0};
#pragma unroll 2
    for (int kt = 0; kt < KT_DFT; ++kt) {
      const _Float16* ap = &wf[m * WF_STRIDE + kt * 32];
      v8h c0 = *(const v8h*)(ap + aoff);        // K = base + aoff + 0..7
      v8h c1 = *(const v8h*)(ap + 16 + aoff);   // K = base + 16 + aoff + 0..7
      v16h a = __builtin_shufflevector(c0, c1, 0, 1, 2, 3, 4, 5, 6, 7,
                                               8, 9, 10, 11, 12, 13, 14, 15);
      const _Float16* bk = dftp + (size_t)kt * NT_TILES * 512 + (size_t)lane * 16;
      v16h b0 = *(const v16h*)(bk + (size_t)nt0 * 512);
      a0 = __builtin_amdgcn_wmma_f32_16x16x32_f16(false, a, false, b0,
                                                  (short)0, a0, false, false);
      v16h b1 = *(const v16h*)(bk + (size_t)nt1 * 512);
      a1 = __builtin_amdgcn_wmma_f32_16x16x32_f16(false, a, false, b1,
                                                  (short)0, a1, false, false);
      v16h b2 = *(const v16h*)(bk + (size_t)nt2 * 512);
      a2 = __builtin_amdgcn_wmma_f32_16x16x32_f16(false, a, false, b2,
                                                  (short)0, a2, false, false);
    }
    // stage this group's tiles to LDS (D layout: row = r + 8*lg, col = lane%16)
#pragma unroll
    for (int r = 0; r < 8; ++r) {
      int rr = (r + 8 * lg) * SPEC_STR + m;
      spec[rr + nt0 * 16] = a0[r];
      spec[rr + nt1 * 16] = a1[r];
      spec[rr + nt2 * 16] = a2[r];
    }
  }
  __syncthreads();

  // ---- mel: fb[64x288] x pow^T[288x16]; wave w owns mel rows 16w..16w+15
  v8f macc = (v8f){0, 0, 0, 0, 0, 0, 0, 0};
#pragma unroll
  for (int kt = 0; kt < KT_MEL; ++kt) {
    v16h afb = *(const v16h*)(fbp + ((size_t)(w * KT_MEL + kt) * 32 + lane) * 16);
    v16h bp;
#pragma unroll
    for (int h = 0; h < 16; ++h) {
      int k = 32 * kt + kmapB(lg, h);           // frequency index
      float pw = 0.0f;
      if (k < NFREQ) {
        float re = spec[m * SPEC_STR + k];
        float im = spec[m * SPEC_STR + COLS_COS + k];
        pw = re * re + im * im;
      }
      bp[h] = (_Float16)pw;
    }
    macc = __builtin_amdgcn_wmma_f32_16x16x32_f16(
        false, afb, false, bp, (short)0, macc, false, false);
  }

#pragma unroll
  for (int r = 0; r < 8; ++r) {
    int mel = 16 * w + r + 8 * lg;
    int t   = t0 + m;                           // <= 1615, always in bounds
    logmel[((size_t)(b * NMEL + mel)) * T_PAD + t] = logf(macc[r] + LOG_GUARD);
  }
}

// ---------------------------------------------------------------------------
// Stage D: masked mean / unbiased std normalization per (b, mel) row.
// ---------------------------------------------------------------------------
__global__ void __launch_bounds__(256) stageD(
    const float* __restrict__ logmel, const int* __restrict__ seq_len,
    float* __restrict__ out)
{
  int row = blockIdx.x;                         // 0..2047
  int b   = row >> 6;
  int mel = row & 63;
  int flen = seq_len[b] / HOP + 1;              // 801..1601

  const float* src = logmel + (size_t)row * T_PAD;
  __shared__ float s1[256], s2[256];
  float sum = 0.0f, sq = 0.0f;
  for (int t = threadIdx.x; t < flen; t += 256) {
    float v = src[t];
    sum += v; sq += v * v;
  }
  s1[threadIdx.x] = sum; s2[threadIdx.x] = sq;
  __syncthreads();
  for (int off = 128; off > 0; off >>= 1) {
    if ((int)threadIdx.x < off) {
      s1[threadIdx.x] += s1[threadIdx.x + off];
      s2[threadIdx.x] += s2[threadIdx.x + off];
    }
    __syncthreads();
  }
  float n    = (float)flen;
  float mean = s1[0] / n;
  float var  = (s2[0] - n * mean * mean) / (n - 1.0f);
  var = var < 0.0f ? 0.0f : var;
  float inv = 1.0f / (sqrtf(var) + 1e-5f);

  float* dst = out + (size_t)row * T_PAD;
  for (int t = threadIdx.x; t < T_PAD; t += 256)
    dst[t] = (t < flen) ? (src[t] - mean) * inv : 0.0f;

  if (mel == 0 && threadIdx.x == 0)
    ((int*)out)[(size_t)NB * NMEL * T_PAD + b] = flen;   // feat_len tail
}

// ---------------------------------------------------------------------------
// Launch. d_in: [0]=x f32[32*256000], [1]=seq_len i32[32],
//               [2]=window f32[512],  [3]=fb f32[64*257]
// ---------------------------------------------------------------------------
extern "C" void kernel_launch(void* const* d_in, const int* in_sizes, int n_in,
                              void* d_out, int out_size, void* d_ws, size_t ws_size,
                              hipStream_t stream) {
  const float* x       = (const float*)d_in[0];
  const int*   seq_len = (const int*)d_in[1];
  const float* win     = (const float*)d_in[2];
  const float* fb      = (const float*)d_in[3];
  float*       out     = (float*)d_out;

  const size_t DFT_SZ     = (size_t)KT_DFT * NT_TILES * 32 * 16 * sizeof(_Float16); // 589824
  const size_t FB_OFF     = (DFT_SZ + 255) & ~(size_t)255;
  const size_t FB_SZ      = (size_t)4 * KT_MEL * 32 * 16 * sizeof(_Float16);        // 36864
  const size_t XP_OFF     = FB_OFF + ((FB_SZ + 255) & ~(size_t)255);
  const size_t XP_SZ      = (size_t)NB * XP_LEN * sizeof(float);                    // ~32.8 MB
  const size_t LOGMEL_OFF = XP_OFF + ((XP_SZ + 255) & ~(size_t)255);
  const size_t LOGMEL_SZ  = (size_t)NB * NMEL * T_PAD * sizeof(float);              // ~13.2 MB
  if (ws_size < LOGMEL_OFF + LOGMEL_SZ) return;   // needs ~47 MB scratch

  char* ws = (char*)d_ws;
  _Float16* dftp   = (_Float16*)(ws);
  _Float16* fbp    = (_Float16*)(ws + FB_OFF);
  float*    xpbuf  = (float*)(ws + XP_OFF);
  float*    logmel = (float*)(ws + LOGMEL_OFF);

  init_dft<<<(KT_DFT * NT_TILES * 32 * 16 + 255) / 256, 256, 0, stream>>>(dftp);
  init_fb<<<(4 * KT_MEL * 32 * 16 + 255) / 256, 256, 0, stream>>>(fb, fbp);
  stageA<<<(int)(((size_t)NB * XP_LEN + 255) / 256), 256, 0, stream>>>(x, xpbuf);
  stageB<<<NB * NTILE_T, 128, 0, stream>>>(xpbuf, win, dftp, fbp, logmel);
  stageD<<<NB * NMEL, 256, 0, stream>>>(logmel, seq_len, out);
}
